// CachedTransformerBlock_79800492360204
// MI455X (gfx1250) — matrix operand order repaired and checked by reading, not compiled, not run on hardware
//
#include <hip/hip_runtime.h>

// ---------------- CDNA5 (gfx1250) wave32 WMMA transformer block ----------------
// D=1024, S=2048, B=4, NH=16, DH=64, HID=4096. All GEMMs in bf16 via
// v_wmma_f32_16x16x32_bf16 (fp32 accumulate). GEMM wave tile: 32(M) x 64(N)
// so every B fragment feeds two WMMAs (8 independent accumulator chains).

typedef unsigned short u16;
typedef __attribute__((ext_vector_type(16))) __bf16 v16bf;
typedef __attribute__((ext_vector_type(8)))  float  v8f;
typedef __attribute__((ext_vector_type(4)))  unsigned int u32x4;
typedef __attribute__((ext_vector_type(2)))  unsigned int u32x2;

#define DIM   1024
#define SEQ   2048
#define BATCH 4
#define TOK   (BATCH * SEQ)   // 8192
#define NHEAD 16
#define DHEAD 64
#define HID   4096
#define QKVN  3072

union Frag16 {
  v16bf v;
  u32x4 q[2];
  u16   u[16];
};

static __device__ __forceinline__ u16 to_bf16(float f) {
  unsigned int u = __float_as_uint(f);
  u += 0x7FFFu + ((u >> 16) & 1u);   // round-to-nearest-even
  return (u16)(u >> 16);
}

static __device__ __forceinline__ v8f v8f_zero() {
  v8f z = {0.f, 0.f, 0.f, 0.f, 0.f, 0.f, 0.f, 0.f};
  return z;
}

static __device__ __forceinline__ v8f wmma_bf16(const Frag16& a, const Frag16& b, v8f c) {
  return __builtin_amdgcn_wmma_f32_16x16x32_bf16(false, a.v, false, b.v, (short)0, c,
                                                 false, false);
}

// A-matrix fragment (16x32 bf16): lanes 0-15 row M=lane hold K = {0..7, 16..23},
// lanes 16-31 row M=lane-16 hold K = {8..15, 24..31}. Two b128 loads per lane.
static __device__ __forceinline__ void load_a_frag(Frag16& f, const u16* row, int kbase,
                                                   bool hi) {
  int ab = kbase + (hi ? 8 : 0);
  f.q[0] = *(const u32x4*)(row + ab);
  f.q[1] = *(const u32x4*)(row + ab + 16);
}

// B-matrix fragment (32x16 bf16) from K-major storage BT[n*ldk + k]:
// lane n = lane&15; lanes 0-15 hold K {0..7, 16..23}, lanes 16-31 hold K {8..15, 24..31}
// (two 16-byte chunks per lane -> two b128 loads).
static __device__ __forceinline__ void load_b_frag(Frag16& f, const u16* bt_row, int kbase,
                                                   bool hi) {
  int kb = kbase + (hi ? 8 : 0);
  f.q[0] = *(const u32x4*)(bt_row + kb);
  f.q[1] = *(const u32x4*)(bt_row + kb + 16);
}

// ---------------- weight convert + transpose: W[K,N] f32 -> WT[N,K] bf16 ------
__global__ __launch_bounds__(256) void k_cvt_t(const float* __restrict__ W,
                                               u16* __restrict__ WT, int K, int N) {
  __shared__ float tile[32][33];
  int n0 = blockIdx.x * 32, k0 = blockIdx.y * 32;
  int tx = threadIdx.x & 31, ty = threadIdx.x >> 5;
#pragma unroll
  for (int i = 0; i < 32; i += 8)
    tile[ty + i][tx] = W[(size_t)(k0 + ty + i) * N + n0 + tx];
  __syncthreads();
#pragma unroll
  for (int i = 0; i < 32; i += 8)
    WT[(size_t)(n0 + ty + i) * K + k0 + tx] = to_bf16(tile[tx][ty + i]);
}

// ---------------- rmsnorm: f32 [T,1024] -> bf16 [T,1024] ----------------------
__global__ __launch_bounds__(256) void k_rmsnorm(const float* __restrict__ x,
                                                 const float* __restrict__ scale,
                                                 u16* __restrict__ out) {
  int t = blockIdx.x;
  const float* row = x + (size_t)t * DIM;
  int base = threadIdx.x * 4;
  float4 v = *(const float4*)(row + base);
  float ss = v.x * v.x + v.y * v.y + v.z * v.z + v.w * v.w;
#pragma unroll
  for (int off = 1; off < 32; off <<= 1) ss += __shfl_xor(ss, off, 32);
  __shared__ float wsum[8];
  if ((threadIdx.x & 31) == 0) wsum[threadIdx.x >> 5] = ss;
  __syncthreads();
  float tot = 0.f;
#pragma unroll
  for (int i = 0; i < 8; ++i) tot += wsum[i];
  float rinv = rsqrtf(tot * (1.0f / DIM) + 1e-6f);
  float4 sc = *(const float4*)(scale + base);
  u32x2 pk;
  pk[0] = (unsigned)to_bf16(v.x * rinv * sc.x) | ((unsigned)to_bf16(v.y * rinv * sc.y) << 16);
  pk[1] = (unsigned)to_bf16(v.z * rinv * sc.z) | ((unsigned)to_bf16(v.w * rinv * sc.w) << 16);
  *(u32x2*)(out + (size_t)t * DIM + base) = pk;
}

// ---------------- GEMM: C_bf16[M,N] = A_bf16[M,K] @ BT_bf16[N,K]^T ------------
// 128 threads = 4 waves; each wave computes a 32(M) x 64(N) tile
// (2 A fragments, 4 B fragments, 8 WMMAs per 32-wide K step).
__global__ __launch_bounds__(128) void k_gemm_bf16(const u16* __restrict__ A,
                                                   const u16* __restrict__ BT,
                                                   u16* __restrict__ C, int N, int K) {
  int wave = threadIdx.x >> 5, lane = threadIdx.x & 31;
  bool hi = lane >= 16;
  int l15 = lane & 15;
  int m0 = blockIdx.y * 32;
  int n0 = blockIdx.x * 256 + wave * 64;
  const u16* arow0 = A + (size_t)(m0 + l15) * K;
  const u16* arow1 = A + (size_t)(m0 + 16 + l15) * K;
  v8f acc[2][4];
#pragma unroll
  for (int mi = 0; mi < 2; ++mi)
#pragma unroll
    for (int t = 0; t < 4; ++t) acc[mi][t] = v8f_zero();
  for (int k0 = 0; k0 < K; k0 += 32) {
    Frag16 a0, a1;
    load_a_frag(a0, arow0, k0, hi);
    load_a_frag(a1, arow1, k0, hi);
#pragma unroll
    for (int t = 0; t < 4; ++t) {
      Frag16 b;
      load_b_frag(b, BT + (size_t)(n0 + t * 16 + l15) * K, k0, hi);
      acc[0][t] = wmma_bf16(a0, b, acc[0][t]);
      acc[1][t] = wmma_bf16(a1, b, acc[1][t]);
    }
  }
#pragma unroll
  for (int mi = 0; mi < 2; ++mi) {
    int mrow = m0 + mi * 16 + (hi ? 8 : 0);
#pragma unroll
    for (int t = 0; t < 4; ++t)
#pragma unroll
      for (int r = 0; r < 8; ++r)
        C[(size_t)(mrow + r) * N + n0 + t * 16 + l15] = to_bf16(acc[mi][t][r]);
  }
}

// ---------------- GEMM + residual: out_f32[M,N] = R[M,N] + A @ BT^T -----------
__global__ __launch_bounds__(128) void k_gemm_resid(const u16* __restrict__ A,
                                                    const u16* __restrict__ BT,
                                                    const float* __restrict__ R,
                                                    float* __restrict__ out, int N, int K) {
  int wave = threadIdx.x >> 5, lane = threadIdx.x & 31;
  bool hi = lane >= 16;
  int l15 = lane & 15;
  int m0 = blockIdx.y * 32;
  int n0 = blockIdx.x * 256 + wave * 64;
  const u16* arow0 = A + (size_t)(m0 + l15) * K;
  const u16* arow1 = A + (size_t)(m0 + 16 + l15) * K;
  v8f acc[2][4];
#pragma unroll
  for (int mi = 0; mi < 2; ++mi)
#pragma unroll
    for (int t = 0; t < 4; ++t) acc[mi][t] = v8f_zero();
  for (int k0 = 0; k0 < K; k0 += 32) {
    Frag16 a0, a1;
    load_a_frag(a0, arow0, k0, hi);
    load_a_frag(a1, arow1, k0, hi);
#pragma unroll
    for (int t = 0; t < 4; ++t) {
      Frag16 b;
      load_b_frag(b, BT + (size_t)(n0 + t * 16 + l15) * K, k0, hi);
      acc[0][t] = wmma_bf16(a0, b, acc[0][t]);
      acc[1][t] = wmma_bf16(a1, b, acc[1][t]);
    }
  }
#pragma unroll
  for (int mi = 0; mi < 2; ++mi) {
    int mrow = m0 + mi * 16 + (hi ? 8 : 0);
#pragma unroll
    for (int t = 0; t < 4; ++t)
#pragma unroll
      for (int r = 0; r < 8; ++r) {
        size_t idx = (size_t)(mrow + r) * N + n0 + t * 16 + l15;
        out[idx] = R[idx] + acc[mi][t][r];
      }
  }
}

// ---------------- fused up/gate GEMM + SwiGLU: h = (A@Wu) * silu(A@Wg) --------
// 32x64 wave tile, two weight matrices: 16 WMMAs per K step from 2 A + 8 B frags.
__global__ __launch_bounds__(128) void k_gemm_swiglu(const u16* __restrict__ A,
                                                     const u16* __restrict__ BTu,
                                                     const u16* __restrict__ BTg,
                                                     u16* __restrict__ H, int N, int K) {
  int wave = threadIdx.x >> 5, lane = threadIdx.x & 31;
  bool hi = lane >= 16;
  int l15 = lane & 15;
  int m0 = blockIdx.y * 32;
  int n0 = blockIdx.x * 256 + wave * 64;
  const u16* arow0 = A + (size_t)(m0 + l15) * K;
  const u16* arow1 = A + (size_t)(m0 + 16 + l15) * K;
  v8f au[2][4], ag[2][4];
#pragma unroll
  for (int mi = 0; mi < 2; ++mi)
#pragma unroll
    for (int t = 0; t < 4; ++t) { au[mi][t] = v8f_zero(); ag[mi][t] = v8f_zero(); }
  for (int k0 = 0; k0 < K; k0 += 32) {
    Frag16 a0, a1;
    load_a_frag(a0, arow0, k0, hi);
    load_a_frag(a1, arow1, k0, hi);
#pragma unroll
    for (int t = 0; t < 4; ++t) {
      size_t brow = (size_t)(n0 + t * 16 + l15) * K;
      Frag16 bu, bg;
      load_b_frag(bu, BTu + brow, k0, hi);
      load_b_frag(bg, BTg + brow, k0, hi);
      au[0][t] = wmma_bf16(a0, bu, au[0][t]);
      au[1][t] = wmma_bf16(a1, bu, au[1][t]);
      ag[0][t] = wmma_bf16(a0, bg, ag[0][t]);
      ag[1][t] = wmma_bf16(a1, bg, ag[1][t]);
    }
  }
#pragma unroll
  for (int mi = 0; mi < 2; ++mi) {
    int mrow = m0 + mi * 16 + (hi ? 8 : 0);
#pragma unroll
    for (int t = 0; t < 4; ++t)
#pragma unroll
      for (int r = 0; r < 8; ++r) {
        float g = ag[mi][t][r];
        float h = au[mi][t][r] * (g / (1.f + __expf(-g)));
        H[(size_t)(mrow + r) * N + n0 + t * 16 + l15] = to_bf16(h);
      }
  }
}

// ---------------- flash attention (causal) ------------------------------------
// qkv_bf16: [TOK, 3072] (q | k | v, each head h at col h*64). One wave per
// (batch, head, 16-query block); 32-key blocks; online softmax; P staged
// through LDS to convert C-layout -> A-layout for the P@V WMMAs.
__global__ __launch_bounds__(128) void k_attn(const u16* __restrict__ qkv,
                                              u16* __restrict__ out) {
  __shared__ u16 plds[4][16 * 32];
  int wave = threadIdx.x >> 5, lane = threadIdx.x & 31;
  bool hi = lane >= 16;
  int l15 = lane & 15;
  int mrow = hi ? 8 : 0;

  int q0 = (blockIdx.x * 4 + wave) * 16;
  int h  = blockIdx.y;
  int b  = blockIdx.z;

  const u16* Qp = qkv + (size_t)b * SEQ * QKVN + h * DHEAD;
  const u16* Kp = Qp + DIM;
  const u16* Vp = Qp + 2 * DIM;

  // Q fragments: A[16 q, 64 d] -> two 16x32 chunks
  Frag16 qa[2];
  const u16* qrow = Qp + (size_t)(q0 + l15) * QKVN;
#pragma unroll
  for (int c = 0; c < 2; ++c) load_a_frag(qa[c], qrow, c * 32, hi);

  float rm[8], rl[8];
#pragma unroll
  for (int r = 0; r < 8; ++r) { rm[r] = -1e30f; rl[r] = 0.f; }
  v8f o[4];
#pragma unroll
  for (int t = 0; t < 4; ++t) o[t] = v8f_zero();

  u16* P = &plds[wave][0];
  const float sc = 0.125f;  // 1/sqrt(64)

  for (int k0 = 0; k0 <= q0 + 15; k0 += 32) {
    // ---- S = Q K^T : two 16x16 tiles (keys k0..k0+15, k0+16..k0+31)
    v8f s[2];
#pragma unroll
    for (int t = 0; t < 2; ++t) s[t] = v8f_zero();
#pragma unroll
    for (int t = 0; t < 2; ++t) {
      const u16* krow = Kp + (size_t)(k0 + t * 16 + l15) * QKVN;
#pragma unroll
      for (int c = 0; c < 2; ++c) {
        Frag16 kf;
        load_b_frag(kf, krow, c * 32, hi);   // B element (kk,n) = K[key=n, d=c*32+kk]
        s[t] = wmma_bf16(qa[c], kf, s[t]);
      }
    }
    // ---- scale + causal mask (C layout: row = r + mrow, col = lane&15)
#pragma unroll
    for (int t = 0; t < 2; ++t)
#pragma unroll
      for (int r = 0; r < 8; ++r) {
        float v = s[t][r] * sc;
        int qq = q0 + r + mrow;
        int kk = k0 + t * 16 + l15;
        s[t][r] = (kk > qq) ? -1e30f : v;
      }
    // ---- online softmax stats (reduce across the 16-lane half)
    float corr[8];
#pragma unroll
    for (int r = 0; r < 8; ++r) {
      float mx = fmaxf(s[0][r], s[1][r]);
#pragma unroll
      for (int off = 1; off < 16; off <<= 1) mx = fmaxf(mx, __shfl_xor(mx, off, 32));
      float mnew = fmaxf(rm[r], mx);
      corr[r] = __expf(rm[r] - mnew);
      rm[r] = mnew;
    }
    // ---- P = exp(S - m); row sums; stage P (bf16) to LDS
#pragma unroll
    for (int r = 0; r < 8; ++r) {
      float p0 = __expf(s[0][r] - rm[r]);
      float p1 = __expf(s[1][r] - rm[r]);
      P[(r + mrow) * 32 + l15]      = to_bf16(p0);
      P[(r + mrow) * 32 + 16 + l15] = to_bf16(p1);
      float rs = p0 + p1;
#pragma unroll
      for (int off = 1; off < 16; off <<= 1) rs += __shfl_xor(rs, off, 32);
      rl[r] = rl[r] * corr[r] + rs;
    }
    // ---- rescale O
#pragma unroll
    for (int t = 0; t < 4; ++t)
#pragma unroll
      for (int r = 0; r < 8; ++r) o[t][r] *= corr[r];

    __builtin_amdgcn_wave_barrier();  // keep LDS store->load ordering

    // ---- reload P in A layout (two contiguous 16B LDS reads per lane)
    Frag16 pf;
    {
      const u16* prow = P + l15 * 32 + mrow;  // mrow doubles as A-layout k-base
      pf.q[0] = *(const u32x4*)(prow);
      pf.q[1] = *(const u32x4*)(prow + 16);
    }
    // ---- O += P @ V  (K-dim = 32 keys; N = 64 dims in 4 tiles)
#pragma unroll
    for (int t = 0; t < 4; ++t) {
      Frag16 vf;
#pragma unroll
      for (int e = 0; e < 16; ++e) {
        int kk = (hi ? 8 : 0) + (e & 7) + (e >> 3) * 16;
        vf.u[e] = Vp[(size_t)(k0 + kk) * QKVN + t * 16 + l15];
      }
      o[t] = wmma_bf16(pf, vf, o[t]);
    }
  }

  // ---- normalize and store bf16 [TOK, DIM]
#pragma unroll
  for (int r = 0; r < 8; ++r) {
    float inv = 1.0f / rl[r];
    size_t tok = (size_t)b * SEQ + q0 + r + mrow;
#pragma unroll
    for (int t = 0; t < 4; ++t)
      out[tok * DIM + h * DHEAD + t * 16 + l15] = to_bf16(o[t][r] * inv);
  }
}

// ---------------- host orchestration ------------------------------------------
extern "C" void kernel_launch(void* const* d_in, const int* in_sizes, int n_in,
                              void* d_out, int out_size, void* d_ws, size_t ws_size,
                              hipStream_t stream) {
  const float* x       = (const float*)d_in[0];
  const float* w_qkv   = (const float*)d_in[1];
  const float* w_o     = (const float*)d_in[2];
  const float* w_up    = (const float*)d_in[3];
  const float* w_gate  = (const float*)d_in[4];
  const float* w_down  = (const float*)d_in[5];
  const float* scale1  = (const float*)d_in[6];
  const float* scale2  = (const float*)d_in[7];

  size_t off = 0;
  auto alloc = [&](size_t bytes) -> char* {
    char* p = (char*)d_ws + off;
    off += (bytes + 255) & ~(size_t)255;
    return p;
  };
  u16*   wqkvT  = (u16*)alloc((size_t)QKVN * DIM * 2);
  u16*   woT    = (u16*)alloc((size_t)DIM * DIM * 2);
  u16*   wupT   = (u16*)alloc((size_t)HID * DIM * 2);
  u16*   wgateT = (u16*)alloc((size_t)HID * DIM * 2);
  u16*   wdownT = (u16*)alloc((size_t)DIM * HID * 2);
  u16*   xn     = (u16*)alloc((size_t)TOK * DIM * 2);
  u16*   qkv    = (u16*)alloc((size_t)TOK * QKVN * 2);
  u16*   attn   = (u16*)alloc((size_t)TOK * DIM * 2);
  float* x1     = (float*)alloc((size_t)TOK * DIM * 4);
  u16*   hbuf   = (u16*)alloc((size_t)TOK * HID * 2);
  (void)ws_size; (void)in_sizes; (void)n_in; (void)out_size;

  // weights -> bf16, transposed to [N,K]
  k_cvt_t<<<dim3(QKVN / 32, DIM / 32), 256, 0, stream>>>(w_qkv, wqkvT, DIM, QKVN);
  k_cvt_t<<<dim3(DIM / 32, DIM / 32), 256, 0, stream>>>(w_o, woT, DIM, DIM);
  k_cvt_t<<<dim3(HID / 32, DIM / 32), 256, 0, stream>>>(w_up, wupT, DIM, HID);
  k_cvt_t<<<dim3(HID / 32, DIM / 32), 256, 0, stream>>>(w_gate, wgateT, DIM, HID);
  k_cvt_t<<<dim3(DIM / 32, HID / 32), 256, 0, stream>>>(w_down, wdownT, HID, DIM);

  // x1 = x + O-proj(attn(rmsnorm(x)))
  k_rmsnorm<<<TOK, 256, 0, stream>>>(x, scale1, xn);
  k_gemm_bf16<<<dim3(QKVN / 256, TOK / 32), 128, 0, stream>>>(xn, wqkvT, qkv, QKVN, DIM);
  k_attn<<<dim3((SEQ / 16) / 4, NHEAD, BATCH), 128, 0, stream>>>(qkv, attn);
  k_gemm_resid<<<dim3(DIM / 256, TOK / 32), 128, 0, stream>>>(attn, woT, x, x1, DIM, DIM);

  // out = x1 + down(swiglu(rmsnorm(x1)))
  k_rmsnorm<<<TOK, 256, 0, stream>>>(x1, scale2, xn);
  k_gemm_swiglu<<<dim3(HID / 256, TOK / 32), 128, 0, stream>>>(xn, wupT, wgateT, hbuf, HID, DIM);
  k_gemm_resid<<<dim3(DIM / 256, TOK / 32), 128, 0, stream>>>(hbuf, wdownT, x1, (float*)d_out,
                                                              DIM, HID);
}
